// Qwen3NextGatedDeltaNet_43963285242743
// MI455X (gfx1250) — matrix operand order repaired and checked by reading, not compiled, not run on hardware
//
#include <hip/hip_runtime.h>
#include <hip/hip_bf16.h>
#include <math.h>

// ---------------------------------------------------------------------------
// Qwen3-Next GatedDeltaNet forward for MI455X (gfx1250, wave32, WMMA + TDM).
//   1. qkvz = x @ W_qkvz           (bf16 WMMA GEMM, TDM-staged tiles)
//   2. ba   = x @ W_ba             (same kernel, N=64)
//   3. causal depthwise conv(K=4) + SiLU over [q,k,v] channels
//   4. per-head RMS-norm + scaling of q,k
//   5. beta = sigmoid(b), g = -exp(A_log)*softplus(a+dt_bias)
//   6. gated delta-rule scan (state 128x128 f32 in registers, 1 block/head)
//   7. gated RMSNorm * SiLU(z)
//   8. out = gated @ W_out         (bf16 WMMA GEMM)
// ---------------------------------------------------------------------------

typedef __bf16 bf16_t;
typedef __attribute__((ext_vector_type(16))) __bf16 v16bf;
typedef __attribute__((ext_vector_type(8)))  float  v8f;
typedef unsigned int u32x4 __attribute__((ext_vector_type(4)));
typedef int          i32x4 __attribute__((ext_vector_type(4)));
typedef int          i32x8 __attribute__((ext_vector_type(8)));

namespace cfg {
constexpr int B = 2, S = 2048, H = 2048;
constexpr int NK = 16, NV = 32, DK = 128, DV = 128;
constexpr int R  = NV / NK;                       // 2
constexpr int KEY_DIM   = NK * DK;                // 2048
constexpr int VALUE_DIM = NV * DV;                // 4096
constexpr int CONV_DIM  = 2 * KEY_DIM + VALUE_DIM;// 8192
constexpr int QKVZ_N    = 2 * KEY_DIM + 2 * VALUE_DIM; // 12288
constexpr int HEAD_STRIDE = QKVZ_N / NK;          // 768 per NK head
constexpr int BA_N = 2 * NV;                      // 64
constexpr int BS   = B * S;                       // 4096
constexpr float EPS = 1e-6f;
}

// ---------------------------------------------------------------------------
// Tensor Data Mover: 2D fp32 tile load global -> LDS, with optional LDS row
// padding. D# packed per CDNA5 ISA §8.3-8.5. Groups 2/3 zero => tile_dim3/4=0
// (unused) => plain 2D tile.
//   pad_code: pad_interval code (4 => every 32 DWORDs, 5 => every 64 DWORDs),
//             pad_amount = 0 => 1 DWORD of padding.
// ---------------------------------------------------------------------------
__device__ __forceinline__ void tdm_load_tile_2d(
    const float* gptr, unsigned lds_byte_off,
    unsigned tile0, unsigned tile1,
    unsigned long long stride0_elems,
    unsigned tensor0, unsigned tensor1,
    unsigned pad_code) {
  unsigned long long ga = (unsigned long long)(size_t)gptr;
  u32x4 g0;
  g0.x = 0x1u;                                             // count=1, user mode
  g0.y = lds_byte_off;                                     // lds_addr
  g0.z = (unsigned)(ga & 0xFFFFFFFFull);                   // global_addr lo
  g0.w = (unsigned)((ga >> 32) & 0x01FFFFFFull) | (2u << 30); // addr hi | type=2
  i32x8 g1;
  unsigned w0 = (2u << 16)                                 // data_size = 4B
              | (1u << 20)                                 // pad_enable
              | (pad_code << 22);                          // pad_interval
  g1[0] = (int)w0;
  g1[1] = (int)((tensor0 & 0xFFFFu) << 16);                // td0[15:0] @ bit48
  g1[2] = (int)(((tensor0 >> 16) & 0xFFFFu) | ((tensor1 & 0xFFFFu) << 16));
  g1[3] = (int)(((tensor1 >> 16) & 0xFFFFu) | (tile0 << 16)); // tile_dim0
  g1[4] = (int)(tile1 & 0xFFFFu);                          // tile_dim1, dim2=0
  g1[5] = (int)(unsigned)(stride0_elems & 0xFFFFFFFFull);  // dim0 stride lo
  g1[6] = (int)(unsigned)((stride0_elems >> 32) & 0xFFFFull);
  g1[7] = 0;
  i32x4 gz = {0, 0, 0, 0};
#if defined(__clang_major__) && (__clang_major__ >= 23)
  i32x8 gz8 = {0, 0, 0, 0, 0, 0, 0, 0};
  __builtin_amdgcn_tensor_load_to_lds(g0, g1, gz, gz, gz8, 0);
#else
  __builtin_amdgcn_tensor_load_to_lds(g0, g1, gz, gz, 0);
#endif
}

// ---------------------------------------------------------------------------
// GEMM: C[M,N] = A[M,K] * B[K,N], f32 in/out, bf16 WMMA, TDM-staged LDS tiles.
// BM=128, BN=64, BK=32. 256 threads = 8 waves (4 M x 2 N), wave tile 32x32.
// Requires M % 128 == 0, N % 64 == 0, K % 32 == 0 (true for all call sites).
// Double-buffered: wave 0 issues next tile pair via TDM, block computes on
// current buffer, s_wait_tensorcnt(2) retires the in-flight pair.
// ---------------------------------------------------------------------------
#define GEMM_BM 128
#define GEMM_BN 64
#define GEMM_BK 32
#define A_TILE_DW (128 * 33)   // 32-dw rows + 1 dw TDM pad -> stride 33
#define B_TILE_DW (32 * 65)    // 64-dw rows + 1 dw TDM pad -> stride 65
#define B_BASE_DW (2 * A_TILE_DW)

__global__ __launch_bounds__(256)
void gemm_f32_bf16wmma(const float* __restrict__ A,
                       const float* __restrict__ Bm,
                       float* __restrict__ C,
                       int M, int N, int K) {
  __shared__ __align__(16) float smem[2 * A_TILE_DW + 2 * B_TILE_DW];

  const int tid  = threadIdx.x;
  const int lane = tid & 31;
  const int wid  = tid >> 5;
  const int wm   = wid & 3;        // 0..3 -> M sub-tiles of 32
  const int wn   = wid >> 2;       // 0..1 -> N sub-tiles of 32
  const int lr   = lane & 15;
  const int hsel = lane >> 4;      // lane half select

  const int m0 = blockIdx.y * GEMM_BM;
  const int n0 = blockIdx.x * GEMM_BN;
  const bool issuer = (wid == 0);  // wave 0 drives the TDM

  v8f acc[2][2] = {};

  const int nk = K / GEMM_BK;
  if (issuer) {   // prologue: stage tile 0 into buffer 0
    tdm_load_tile_2d(A + (size_t)m0 * K, 0u, GEMM_BK, GEMM_BM,
                     (unsigned long long)K, (unsigned)K, (unsigned)M, 4u);
    tdm_load_tile_2d(Bm + n0, (unsigned)(B_BASE_DW * 4), GEMM_BN, GEMM_BK,
                     (unsigned long long)N, (unsigned)N, (unsigned)K, 5u);
  }

  for (int i = 0; i < nk; ++i) {
    const int cur = i & 1;
    const int nxt = cur ^ 1;
    if (issuer) {
      if (i + 1 < nk) {
        const int k1 = (i + 1) * GEMM_BK;
        tdm_load_tile_2d(A + (size_t)m0 * K + k1,
                         (unsigned)nxt * (unsigned)(A_TILE_DW * 4),
                         GEMM_BK, GEMM_BM,
                         (unsigned long long)K, (unsigned)K, (unsigned)M, 4u);
        tdm_load_tile_2d(Bm + (size_t)k1 * N + n0,
                         (unsigned)(B_BASE_DW * 4) + (unsigned)nxt * (unsigned)(B_TILE_DW * 4),
                         GEMM_BN, GEMM_BK,
                         (unsigned long long)N, (unsigned)N, (unsigned)K, 5u);
        __builtin_amdgcn_s_wait_tensorcnt(2);   // tile i landed, pair i+1 in flight
      } else {
        __builtin_amdgcn_s_wait_tensorcnt(0);   // last tile landed
      }
    }
    __syncthreads();   // publish buffer `cur` to all waves

    // ---- fragment builds per ISA 7.12.2 layouts (f32 LDS -> bf16 regs) ----
    const float* At = smem + cur * A_TILE_DW;
    const float* Bt = smem + B_BASE_DW + cur * B_TILE_DW;
    v16bf afr[2], bfr[2];
    #pragma unroll
    for (int mi = 0; mi < 2; ++mi) {
      const int row = wm * 32 + mi * 16 + lr;
      const int base = row * 33;
      #pragma unroll
      for (int e = 0; e < 16; ++e) {
        const int k = ((e < 8) ? e : e + 8) + hsel * 8;   // A 16x32 layout
        afr[mi][e] = (bf16_t)At[base + k];
      }
    }
    #pragma unroll
    for (int ni = 0; ni < 2; ++ni) {
      const int col = wn * 32 + ni * 16 + lr;
      const int kb  = hsel * 8;                           // B 32x16 layout
      #pragma unroll
      for (int e = 0; e < 16; ++e)
        bfr[ni][e] = (bf16_t)Bt[(((e < 8) ? e : e + 8) + kb) * 65 + col];
    }

    #pragma unroll
    for (int mi = 0; mi < 2; ++mi)
      #pragma unroll
      for (int ni = 0; ni < 2; ++ni)
        acc[mi][ni] = __builtin_amdgcn_wmma_f32_16x16x32_bf16(
            false, afr[mi], false, bfr[ni], (short)0, acc[mi][ni], false, false);

    __syncthreads();   // all reads of `cur` done before TDM reuses it
  }

  // ---- store C (16x16 f32 C/D layout: VGPR r -> M = 8*hsel + r) ----------
  #pragma unroll
  for (int mi = 0; mi < 2; ++mi) {
    #pragma unroll
    for (int ni = 0; ni < 2; ++ni) {
      const int col = n0 + wn * 32 + ni * 16 + lr;
      #pragma unroll
      for (int r = 0; r < 8; ++r) {
        const int row = m0 + wm * 32 + mi * 16 + hsel * 8 + r;
        C[(size_t)row * N + col] = acc[mi][ni][r];
      }
    }
  }
}

// ---------------------------------------------------------------------------
// Causal depthwise conv (K_CONV=4) + SiLU; channels gathered from qkvz.
// ---------------------------------------------------------------------------
__global__ void conv_silu_kernel(const float* __restrict__ qkvz,
                                 const float* __restrict__ conv_w,
                                 float* __restrict__ qc,
                                 float* __restrict__ kc,
                                 float* __restrict__ vc) {
  using namespace cfg;
  size_t idx = (size_t)blockIdx.x * blockDim.x + threadIdx.x;
  if (idx >= (size_t)BS * CONV_DIM) return;
  int c  = (int)(idx % CONV_DIM);
  int bs = (int)(idx / CONV_DIM);
  int s  = bs % S;
  int b  = bs / S;

  int col;
  if (c < KEY_DIM) {                      // q block
    col = (c >> 7) * HEAD_STRIDE + (c & 127);
  } else if (c < 2 * KEY_DIM) {           // k block
    int c2 = c - KEY_DIM;
    col = (c2 >> 7) * HEAD_STRIDE + DK + (c2 & 127);
  } else {                                // v block: [NK, R*DV] region
    int c3 = c - 2 * KEY_DIM;
    col = (c3 >> 8) * HEAD_STRIDE + 2 * DK + (c3 & 255);
  }

  const float* w = conv_w + (size_t)c * 4;
  float acc = 0.f;
  #pragma unroll
  for (int j = 0; j < 4; ++j) {
    int sp = s - 3 + j;
    if (sp >= 0)
      acc += qkvz[((size_t)b * S + sp) * QKVZ_N + col] * w[j];
  }
  float r = acc / (1.f + __expf(-acc));   // SiLU

  size_t row = (size_t)bs;
  if (c < KEY_DIM)            qc[row * KEY_DIM   + c]                 = r;
  else if (c < 2 * KEY_DIM)   kc[row * KEY_DIM   + (c - KEY_DIM)]     = r;
  else                        vc[row * VALUE_DIM + (c - 2 * KEY_DIM)] = r;
}

// ---------------------------------------------------------------------------
// Per-head RMS norm + scaling of q,k. One block (128 thr) per (b,s,hk).
// ---------------------------------------------------------------------------
__global__ __launch_bounds__(128)
void qk_norm_kernel(float* __restrict__ qc, float* __restrict__ kc) {
  using namespace cfg;
  __shared__ float red[128];
  int rowid = blockIdx.x;           // b*S*NK
  int hk = rowid % NK;
  int bs = rowid / NK;
  int d  = threadIdx.x;
  size_t addr = (size_t)bs * KEY_DIM + hk * DK + d;

  float qv = qc[addr];
  float kv = kc[addr];

  red[d] = qv * qv;
  __syncthreads();
  for (int st = 64; st > 0; st >>= 1) {
    if (d < st) red[d] += red[d + st];
    __syncthreads();
  }
  float qinv = rsqrtf(red[0] * (1.f / DK) + EPS);
  __syncthreads();

  red[d] = kv * kv;
  __syncthreads();
  for (int st = 64; st > 0; st >>= 1) {
    if (d < st) red[d] += red[d + st];
    __syncthreads();
  }
  float kinv = rsqrtf(red[0] * (1.f / DK) + EPS);

  qc[addr] = qv * qinv * (1.f / 128.f);              // inv_scale^2 = 1/DK
  kc[addr] = kv * kinv * 0.08838834764831845f;       // inv_scale = DK^-0.5
}

// ---------------------------------------------------------------------------
// beta = sigmoid(b_gate); g = -exp(A_log) * softplus(a_gate + dt_bias)
// ---------------------------------------------------------------------------
__global__ void gate_kernel(const float* __restrict__ ba,
                            const float* __restrict__ dt_bias,
                            const float* __restrict__ A_log,
                            float* __restrict__ gbuf,
                            float* __restrict__ betabuf) {
  using namespace cfg;
  int idx = blockIdx.x * blockDim.x + threadIdx.x;
  if (idx >= BS * NV) return;
  int hv = idx % NV;
  int bs = idx / NV;
  int hk = hv / R;
  int r  = hv % R;
  const float* bap = ba + (size_t)bs * BA_N + hk * (2 * R);
  float bg = bap[r];
  float ag = bap[R + r];
  float beta = 1.f / (1.f + __expf(-bg));
  float x = ag + dt_bias[hv];
  float sp = (x > 20.f) ? x : log1pf(__expf(x));     // softplus
  gbuf[idx]    = -__expf(A_log[hv]) * sp;
  betabuf[idx] = beta;
}

// ---------------------------------------------------------------------------
// Gated delta-rule scan. One block per (b,hv). 256 threads.
// State S[128][128] f32: thread (khalf=tid>>7, vcol=tid&127) owns
// S[khalf*64 .. +63][vcol] in 64 registers.
// ---------------------------------------------------------------------------
__global__ __launch_bounds__(256)
void delta_scan_kernel(const float* __restrict__ qc,
                       const float* __restrict__ kc,
                       const float* __restrict__ vc,
                       const float* __restrict__ gbuf,
                       const float* __restrict__ betabuf,
                       float* __restrict__ core) {
  using namespace cfg;
  int bh = blockIdx.x;
  int hv = bh % NV;
  int b  = bh / NV;
  int hk = hv / R;
  int tid   = threadIdx.x;
  int vcol  = tid & 127;
  int khalf = tid >> 7;

  float St[64];
  #pragma unroll
  for (int i = 0; i < 64; ++i) St[i] = 0.f;

  __shared__ float sq[128], sk[128], sv[128], sdelta[128], spart[256];

  for (int t = 0; t < S; ++t) {
    size_t bs = (size_t)b * S + t;
    if (tid < 128) {
      sq[tid] = qc[bs * KEY_DIM + hk * DK + tid];
      sk[tid] = kc[bs * KEY_DIM + hk * DK + tid];
    } else {
      int d = tid - 128;
      sv[d] = vc[bs * VALUE_DIM + hv * DV + d];
    }
    float gt = gbuf[bs * NV + hv];
    float bt = betabuf[bs * NV + hv];
    __syncthreads();

    float eg  = __expf(gt);
    float acc = 0.f;
    #pragma unroll
    for (int kk = 0; kk < 64; ++kk) {
      float s = St[kk] * eg;            // state decay
      St[kk] = s;
      acc += sk[khalf * 64 + kk] * s;   // kv partial
    }
    spart[tid] = acc;
    __syncthreads();

    if (khalf == 0) {
      float kv = spart[vcol] + spart[vcol + 128];
      sdelta[vcol] = (sv[vcol] - kv) * bt;
    }
    __syncthreads();

    float dl   = sdelta[vcol];
    float acco = 0.f;
    #pragma unroll
    for (int kk = 0; kk < 64; ++kk) {
      int k = khalf * 64 + kk;
      float s = St[kk] + sk[k] * dl;    // rank-1 update
      St[kk] = s;
      acco += sq[k] * s;                // output partial
    }
    spart[tid] = acco;
    __syncthreads();

    if (khalf == 0)
      core[(bs * NV + hv) * DV + vcol] = spart[vcol] + spart[vcol + 128];
    __syncthreads();
  }
}

// ---------------------------------------------------------------------------
// gated = SiLU(z) * (rms_norm(core) * norm_w). One block (128) per (b,s,hv).
// ---------------------------------------------------------------------------
__global__ __launch_bounds__(128)
void gated_norm_kernel(const float* __restrict__ core,
                       const float* __restrict__ qkvz,
                       const float* __restrict__ norm_w,
                       float* __restrict__ gated) {
  using namespace cfg;
  __shared__ float red[128];
  int rowid = blockIdx.x;       // b*S*NV
  int hv = rowid % NV;
  int bs = rowid / NV;
  int d  = threadIdx.x;

  float o = core[((size_t)bs * NV + hv) * DV + d];
  red[d] = o * o;
  __syncthreads();
  for (int st = 64; st > 0; st >>= 1) {
    if (d < st) red[d] += red[d + st];
    __syncthreads();
  }
  float inv = rsqrtf(red[0] * (1.f / DV) + EPS);
  float xn  = o * inv * norm_w[d];

  int hk = hv / R, r = hv % R;
  float z = qkvz[(size_t)bs * QKVZ_N + hk * HEAD_STRIDE + 2 * DK + R * DV + r * DV + d];
  float sz = z / (1.f + __expf(-z));    // SiLU(z)
  gated[(size_t)bs * VALUE_DIM + hv * DV + d] = sz * xn;
}

// ---------------------------------------------------------------------------
// Host-side launcher
// ---------------------------------------------------------------------------
extern "C" void kernel_launch(void* const* d_in, const int* in_sizes, int n_in,
                              void* d_out, int out_size, void* d_ws, size_t ws_size,
                              hipStream_t stream) {
  using namespace cfg;
  const float* x       = (const float*)d_in[0];
  const float* W_qkvz  = (const float*)d_in[1];
  const float* W_ba    = (const float*)d_in[2];
  const float* conv_w  = (const float*)d_in[3];
  const float* dt_bias = (const float*)d_in[4];
  const float* A_log   = (const float*)d_in[5];
  const float* norm_w  = (const float*)d_in[6];
  const float* W_out   = (const float*)d_in[7];
  float* out = (float*)d_out;

  // workspace carve (f32 elements)
  float* ws = (float*)d_ws;
  size_t off = 0;
  float* qkvz = ws + off; off += (size_t)BS * QKVZ_N;     // 50.3M
  float* ba   = ws + off; off += (size_t)BS * BA_N;       // 0.26M
  float* qc   = ws + off; off += (size_t)BS * KEY_DIM;    // 8.4M
  float* kc   = ws + off; off += (size_t)BS * KEY_DIM;    // 8.4M
  float* vc   = ws + off; off += (size_t)BS * VALUE_DIM;  // 16.8M
  float* gbuf = ws + off; off += (size_t)BS * NV;
  float* betb = ws + off; off += (size_t)BS * NV;
  float* core = ws + off; off += (size_t)BS * NV * DV;    // 16.8M
  float* gated = vc;   // vc dead after the scan -> alias

  // 1) qkvz projection: [4096,2048] x [2048,12288]
  {
    dim3 grid(QKVZ_N / GEMM_BN, BS / GEMM_BM);
    gemm_f32_bf16wmma<<<grid, 256, 0, stream>>>(x, W_qkvz, qkvz, BS, QKVZ_N, H);
  }
  // 2) ba projection: [4096,2048] x [2048,64]
  {
    dim3 grid(BA_N / GEMM_BN, BS / GEMM_BM);
    gemm_f32_bf16wmma<<<grid, 256, 0, stream>>>(x, W_ba, ba, BS, BA_N, H);
  }
  // 3) depthwise conv + SiLU
  {
    size_t total = (size_t)BS * CONV_DIM;
    conv_silu_kernel<<<(unsigned)((total + 255) / 256), 256, 0, stream>>>(
        qkvz, conv_w, qc, kc, vc);
  }
  // 4) q/k RMS norm + scale
  qk_norm_kernel<<<BS * NK, 128, 0, stream>>>(qc, kc);
  // 5) gates
  gate_kernel<<<(BS * NV + 255) / 256, 256, 0, stream>>>(ba, dt_bias, A_log, gbuf, betb);
  // 6) gated delta-rule scan
  delta_scan_kernel<<<B * NV, 256, 0, stream>>>(qc, kc, vc, gbuf, betb, core);
  // 7) gated RMS norm * SiLU(z)
  gated_norm_kernel<<<BS * NV, 128, 0, stream>>>(core, qkvz, norm_w, gated);
  // 8) output projection: [4096,4096] x [4096,2048]
  {
    dim3 grid(H / GEMM_BN, BS / GEMM_BM);
    gemm_f32_bf16wmma<<<grid, 256, 0, stream>>>(gated, W_out, out, BS, H, VALUE_DIM);
  }
}
